// Attention_6640019439812
// MI455X (gfx1250) — matrix operand order repaired and checked by reading, not compiled, not run on hardware
//
#include <hip/hip_runtime.h>
#include <stdint.h>

// ---------------- types ----------------
typedef __attribute__((ext_vector_type(16))) __bf16          v16bf;
typedef __attribute__((ext_vector_type(8)))  float           v8f;
typedef __attribute__((ext_vector_type(8)))  unsigned short  us8;
typedef __attribute__((ext_vector_type(16))) unsigned short  us16;
typedef __attribute__((__vector_size__(16))) int             i32x4;

#define NSEQ 2048
#define DMODEL 1024
#define DHEAD 64
#define NHEAD 16

// fp32 -> bf16 (round to nearest even)
__device__ __forceinline__ unsigned short f2bf(float f) {
  union { float f; unsigned u; } v; v.f = f;
  unsigned r = v.u + 0x7FFFu + ((v.u >> 16) & 1u);
  return (unsigned short)(r >> 16);
}

// A-matrix fragment (16x32 bf16): per lane, 8 contiguous at p, 8 contiguous at p+16
__device__ __forceinline__ v16bf frag_a(const unsigned short* p) {
  us8 lo = *(const us8*)p;
  us8 hi = *(const us8*)(p + 16);
  us16 t = __builtin_shufflevector(lo, hi, 0,1,2,3,4,5,6,7,8,9,10,11,12,13,14,15);
  return __builtin_bit_cast(v16bf, t);
}
// B-matrix fragment (32x16 bf16): per lane, 8 contiguous at p, 8 contiguous at p+16
__device__ __forceinline__ v16bf frag_b(const unsigned short* p) {
  us8 lo = *(const us8*)p;
  us8 hi = *(const us8*)(p + 16);
  us16 t = __builtin_shufflevector(lo, hi, 0,1,2,3,4,5,6,7,8,9,10,11,12,13,14,15);
  return __builtin_bit_cast(v16bf, t);
}

__device__ __forceinline__ v8f wmma_bf16(v16bf a, v16bf b, v8f c) {
  return __builtin_amdgcn_wmma_f32_16x16x32_bf16(false, a, false, b, (short)0, c,
                                                 false, false);
}

// ---------------- async global->LDS copy (CDNA5) ----------------
#if defined(__has_builtin)
#if __has_builtin(__builtin_amdgcn_global_load_async_to_lds_b128) && \
    __has_builtin(__builtin_amdgcn_s_wait_asynccnt)
#define HAVE_ASYNC 1
#endif
#endif

__device__ __forceinline__ void cp_b128(const unsigned short* g, unsigned short* l) {
#ifdef HAVE_ASYNC
  __builtin_amdgcn_global_load_async_to_lds_b128((i32x4*)g, (i32x4*)l, 0, 0);
#else
  *(int4*)l = *(const int4*)g;
#endif
}
__device__ __forceinline__ void cp_wait() {
#ifdef HAVE_ASYNC
  __builtin_amdgcn_s_wait_asynccnt(0);
#endif
}

// ---------------- cast kernels ----------------
__global__ __launch_bounds__(256) void k_cast(const float* __restrict__ src,
                                              unsigned short* __restrict__ dst, int n) {
  int i = blockIdx.x * 256 + threadIdx.x;
  if (i < n) dst[i] = f2bf(src[i]);
}
// W [K x N] row-major -> Wt [N x K] row-major, bf16
__global__ __launch_bounds__(256) void k_cast_t(const float* __restrict__ W,
                                                unsigned short* __restrict__ Wt,
                                                int K, int N) {
  int i = blockIdx.x * 256 + threadIdx.x;
  if (i < K * N) {
    int k = i / N, n = i % N;
    Wt[(size_t)n * K + k] = f2bf(W[i]);
  }
}

// ---------------- fused QKV projection GEMM ----------------
// C[m, c] = sum_k xb[m,k] * W[k,c], c in [0,3072): [Wq | Wkv]
// Scatter: Q[b,h,n,d], K[b,h,n,d], Vt[b,h,d,n] (bf16)
__global__ __launch_bounds__(128) void k_gemm_qkv(
    const unsigned short* __restrict__ xb,
    const unsigned short* __restrict__ WqT,
    const unsigned short* __restrict__ WkvT,
    unsigned short* __restrict__ Q,
    unsigned short* __restrict__ K,
    unsigned short* __restrict__ Vt) {
  const int KD = DMODEL;
  __shared__ unsigned short As[64 * 40];
  __shared__ unsigned short Bs[64 * 40];
  int m0 = blockIdx.x * 64;
  int c0 = blockIdx.y * 64;
  const unsigned short* Bt = (c0 < 1024) ? (WqT + (size_t)c0 * KD)
                                         : (WkvT + (size_t)(c0 - 1024) * KD);
  const unsigned short* Ab = xb + (size_t)m0 * KD;

  int tid = threadIdx.x;
  int wave = tid >> 5, lane = tid & 31;
  int lr = lane & 15, lh = lane >> 4;

  v8f zero = {0.f, 0.f, 0.f, 0.f, 0.f, 0.f, 0.f, 0.f};
  v8f acc[4] = {zero, zero, zero, zero};

  for (int kk = 0; kk < KD; kk += 32) {
#pragma unroll
    for (int i = 0; i < 2; i++) {
      int c = tid + i * 128;          // 0..255
      int row = c >> 2;               // 0..63
      int off = (c & 3) * 8;          // 0,8,16,24 (elems)
      cp_b128(Ab + (size_t)row * KD + kk + off, &As[row * 40 + off]);
      cp_b128(Bt + (size_t)row * KD + kk + off, &Bs[row * 40 + off]);
    }
    cp_wait();
    __syncthreads();
    v16bf a = frag_a(&As[(wave * 16 + lr) * 40 + lh * 8]);
#pragma unroll
    for (int nt = 0; nt < 4; nt++) {
      v16bf b = frag_b(&Bs[(nt * 16 + lr) * 40 + lh * 8]);
      acc[nt] = wmma_bf16(a, b, acc[nt]);
    }
    __syncthreads();
  }

  int region = c0 >> 10;            // 0:Q 1:K 2:V
  int h = (c0 & 1023) >> 6;
  int bI = m0 >> 11;
  int n0 = m0 & 2047;
  int bh = bI * NHEAD + h;
#pragma unroll
  for (int nt = 0; nt < 4; nt++) {
    int dd = nt * 16 + lr;
#pragma unroll
    for (int r = 0; r < 8; r++) {
      int m = n0 + wave * 16 + r + lh * 8;     // seq index
      unsigned short val = f2bf(acc[nt][r]);
      if (region == 0)
        Q[((size_t)bh * NSEQ + m) * DHEAD + dd] = val;
      else if (region == 1)
        K[((size_t)bh * NSEQ + m) * DHEAD + dd] = val;
      else
        Vt[((size_t)bh * DHEAD + dd) * NSEQ + m] = val;
    }
  }
}

// ---------------- flash attention (causal) ----------------
// grid (32 qblocks, 32 b*h), 128 threads = 4 waves, 16 query rows per wave
__global__ __launch_bounds__(128) void k_attn(
    const unsigned short* __restrict__ Q,
    const unsigned short* __restrict__ K,
    const unsigned short* __restrict__ Vt,
    unsigned short* __restrict__ O) {
  __shared__ unsigned short Ks[64 * 72];      // [key j][d]
  __shared__ unsigned short Vs[64 * 72];      // [d][key j]  (V transposed tile)
  __shared__ unsigned short Ps[4 * 16 * 72];  // per-wave P staging

  int qb = blockIdx.x;
  int bh = blockIdx.y;
  int tid = threadIdx.x, wave = tid >> 5, lane = tid & 31;
  int lr = lane & 15, lh = lane >> 4;
  int row0 = qb * 64;

  const unsigned short* Qg = Q + (size_t)bh * NSEQ * DHEAD;
  const unsigned short* Kg = K + (size_t)bh * NSEQ * DHEAD;
  const unsigned short* Vg = Vt + (size_t)bh * DHEAD * NSEQ;

  int qrow = row0 + wave * 16 + lr;
  v16bf qf[2];
  qf[0] = frag_a(Qg + (size_t)qrow * DHEAD + 0 + lh * 8);
  qf[1] = frag_a(Qg + (size_t)qrow * DHEAD + 32 + lh * 8);

  v8f zero = {0.f, 0.f, 0.f, 0.f, 0.f, 0.f, 0.f, 0.f};
  v8f o[4] = {zero, zero, zero, zero};
  float mrow[8], lsum[8];
#pragma unroll
  for (int r = 0; r < 8; r++) { mrow[r] = -3.0e38f; lsum[r] = 0.f; }

  const float scale = 0.125f;  // 64^-0.5

  for (int jb = 0; jb <= qb; jb++) {
    // stage K tile [64 x 64] and V^T tile [64 x 64]
#pragma unroll
    for (int i = 0; i < 4; i++) {
      int c = tid + i * 128;            // 0..511
      int row = c >> 3;                 // 0..63
      int off = (c & 7) * 8;            // 0..56 (elems)
      cp_b128(Kg + (size_t)(jb * 64 + row) * DHEAD + off, &Ks[row * 72 + off]);
      cp_b128(Vg + (size_t)row * NSEQ + jb * 64 + off, &Vs[row * 72 + off]);
    }
    cp_wait();
    __syncthreads();

    // S = (Q K^T) * scale, causal mask
    float s[4][8];
#pragma unroll
    for (int jt = 0; jt < 4; jt++) {
      v8f sa = zero;
#pragma unroll
      for (int kc = 0; kc < 2; kc++) {
        v16bf b = frag_b(&Ks[(jt * 16 + lr) * 72 + kc * 32 + lh * 8]);
        sa = wmma_bf16(qf[kc], b, sa);
      }
      int j = jb * 64 + jt * 16 + lr;
#pragma unroll
      for (int r = 0; r < 8; r++) {
        int irow = row0 + wave * 16 + r + lh * 8;
        s[jt][r] = (j > irow) ? -3.0e38f : sa[r] * scale;
      }
    }

    // online softmax (rows split as r + 8*lh, cols across 16 lanes x 4 tiles)
#pragma unroll
    for (int r = 0; r < 8; r++) {
      float rm = fmaxf(fmaxf(s[0][r], s[1][r]), fmaxf(s[2][r], s[3][r]));
      rm = fmaxf(rm, __shfl_xor(rm, 1, 32));
      rm = fmaxf(rm, __shfl_xor(rm, 2, 32));
      rm = fmaxf(rm, __shfl_xor(rm, 4, 32));
      rm = fmaxf(rm, __shfl_xor(rm, 8, 32));
      float mnew = fmaxf(mrow[r], rm);
      float corr = __expf(mrow[r] - mnew);
      mrow[r] = mnew;
      float rs = 0.f;
#pragma unroll
      for (int jt = 0; jt < 4; jt++) {
        float p = __expf(s[jt][r] - mnew);
        s[jt][r] = p;
        rs += p;
      }
      rs += __shfl_xor(rs, 1, 32);
      rs += __shfl_xor(rs, 2, 32);
      rs += __shfl_xor(rs, 4, 32);
      rs += __shfl_xor(rs, 8, 32);
      lsum[r] = lsum[r] * corr + rs;
#pragma unroll
      for (int dt = 0; dt < 4; dt++) o[dt][r] *= corr;
    }

    // C-layout -> A-layout transpose of P through per-wave LDS region
    unsigned short* Pw = &Ps[wave * 16 * 72];
#pragma unroll
    for (int jt = 0; jt < 4; jt++)
#pragma unroll
      for (int r = 0; r < 8; r++)
        Pw[(r + lh * 8) * 72 + jt * 16 + lr] = f2bf(s[jt][r]);
    asm volatile("s_wait_dscnt 0" ::: "memory");  // DS in-order within wave

    v16bf pf[2];
    pf[0] = frag_a(Pw + (size_t)lr * 72 + 0 + lh * 8);
    pf[1] = frag_a(Pw + (size_t)lr * 72 + 32 + lh * 8);

    // O += P * V  (B-fragment = V^T tile rows: fixed dd, contiguous j)
#pragma unroll
    for (int dt = 0; dt < 4; dt++)
#pragma unroll
      for (int kc = 0; kc < 2; kc++) {
        v16bf b = frag_b(&Vs[(dt * 16 + lr) * 72 + kc * 32 + lh * 8]);
        o[dt] = wmma_bf16(pf[kc], b, o[dt]);
      }
    __syncthreads();
  }

  // normalize + store O as [b, n, h*d] bf16
  int bI = bh >> 4, h = bh & 15;
#pragma unroll
  for (int dt = 0; dt < 4; dt++)
#pragma unroll
    for (int r = 0; r < 8; r++) {
      int irow = row0 + wave * 16 + r + lh * 8;
      float val = o[dt][r] / lsum[r];
      O[((size_t)bI * NSEQ + irow) * DMODEL + h * DHEAD + dt * 16 + lr] = f2bf(val);
    }
}

// ---------------- output projection GEMM (+bias, fp32 out) ----------------
__global__ __launch_bounds__(128) void k_gemm_out(
    const unsigned short* __restrict__ A,
    const unsigned short* __restrict__ Bt,
    const float* __restrict__ bias,
    float* __restrict__ Cout) {
  const int KD = DMODEL;
  __shared__ unsigned short As[64 * 40];
  __shared__ unsigned short Bs[64 * 40];
  int m0 = blockIdx.x * 64;
  int c0 = blockIdx.y * 64;
  const unsigned short* Ab = A + (size_t)m0 * KD;
  const unsigned short* Bb = Bt + (size_t)c0 * KD;

  int tid = threadIdx.x;
  int wave = tid >> 5, lane = tid & 31;
  int lr = lane & 15, lh = lane >> 4;

  v8f zero = {0.f, 0.f, 0.f, 0.f, 0.f, 0.f, 0.f, 0.f};
  v8f acc[4] = {zero, zero, zero, zero};

  for (int kk = 0; kk < KD; kk += 32) {
#pragma unroll
    for (int i = 0; i < 2; i++) {
      int c = tid + i * 128;
      int row = c >> 2;
      int off = (c & 3) * 8;
      cp_b128(Ab + (size_t)row * KD + kk + off, &As[row * 40 + off]);
      cp_b128(Bb + (size_t)row * KD + kk + off, &Bs[row * 40 + off]);
    }
    cp_wait();
    __syncthreads();
    v16bf a = frag_a(&As[(wave * 16 + lr) * 40 + lh * 8]);
#pragma unroll
    for (int nt = 0; nt < 4; nt++) {
      v16bf b = frag_b(&Bs[(nt * 16 + lr) * 40 + lh * 8]);
      acc[nt] = wmma_bf16(a, b, acc[nt]);
    }
    __syncthreads();
  }

#pragma unroll
  for (int nt = 0; nt < 4; nt++) {
    int c = c0 + nt * 16 + lr;
    float bv = bias[c];
#pragma unroll
    for (int r = 0; r < 8; r++) {
      int m = m0 + wave * 16 + r + lh * 8;
      Cout[(size_t)m * DMODEL + c] = acc[nt][r] + bv;
    }
  }
}

// ---------------- launch ----------------
extern "C" void kernel_launch(void* const* d_in, const int* in_sizes, int n_in,
                              void* d_out, int out_size, void* d_ws, size_t ws_size,
                              hipStream_t stream) {
  const float* x   = (const float*)d_in[0];
  const float* Wq  = (const float*)d_in[1];
  const float* Wkv = (const float*)d_in[2];
  const float* Wo  = (const float*)d_in[3];
  const float* bo  = (const float*)d_in[4];
  float* out = (float*)d_out;

  char* ws = (char*)d_ws;
  const size_t MB = 1024 * 1024;
  unsigned short* xb   = (unsigned short*)(ws + 0);        // 8 MB
  unsigned short* WqT  = (unsigned short*)(ws + 8 * MB);   // 2 MB
  unsigned short* WkvT = (unsigned short*)(ws + 10 * MB);  // 4 MB
  unsigned short* WoT  = (unsigned short*)(ws + 14 * MB);  // 2 MB
  unsigned short* Qb   = (unsigned short*)(ws + 16 * MB);  // 8 MB
  unsigned short* Kb   = (unsigned short*)(ws + 24 * MB);  // 8 MB
  unsigned short* Vtb  = (unsigned short*)(ws + 32 * MB);  // 8 MB (transposed V)
  unsigned short* Ob   = (unsigned short*)(ws + 40 * MB);  // 8 MB

  int nx = 2 * NSEQ * DMODEL;
  k_cast<<<(nx + 255) / 256, 256, 0, stream>>>(x, xb, nx);
  k_cast_t<<<(1024 * 1024 + 255) / 256, 256, 0, stream>>>(Wq, WqT, 1024, 1024);
  k_cast_t<<<(1024 * 2048 + 255) / 256, 256, 0, stream>>>(Wkv, WkvT, 1024, 2048);
  k_cast_t<<<(1024 * 1024 + 255) / 256, 256, 0, stream>>>(Wo, WoT, 1024, 1024);

  k_gemm_qkv<<<dim3(64, 48), 128, 0, stream>>>(xb, WqT, WkvT, Qb, Kb, Vtb);
  k_attn<<<dim3(32, 32), 128, 0, stream>>>(Qb, Kb, Vtb, Ob);
  k_gemm_out<<<dim3(64, 16), 128, 0, stream>>>(Ob, WoT, bo, out);
}
